// SpatialNetLayer_85220741087896
// MI455X (gfx1250) — matrix operand order repaired and checked by reading, not compiled, not run on hardware
//
#include <hip/hip_runtime.h>
#include <hip/hip_bf16.h>
#include <math.h>

typedef __attribute__((ext_vector_type(16))) __bf16 v16bf;
typedef __attribute__((ext_vector_type(8)))  __bf16 v8bf;
typedef __attribute__((ext_vector_type(8)))  float  v8f;

#define B_   2
#define F_   129
#define T_   250
#define H_   96
#define S_   8
#define CPG_ 12          // channels per conv group (96/8)
#define KW_  5
#define DI_  192         // mamba inner dim
#define DS_  16          // state dim
#define DR_  6           // dt rank
#define KC_  4           // mamba conv width
#define NSEQ   (B_*F_)        // 258
#define MROWS  (NSEQ*T_)      // 64500
#define MPAD   64512          // ceil to multiple of 32
#define DBLS   64             // padded row stride of dbl (38 -> 64)
#define EPS_   1e-5f

__device__ __forceinline__ __bf16 f2bf(float f) {
  unsigned u = __builtin_bit_cast(unsigned, f);
  unsigned r = u + 0x7FFFu + ((u >> 16) & 1u);   // round-to-nearest-even
  unsigned short h = (unsigned short)(r >> 16);
  return __builtin_bit_cast(__bf16, h);
}
__device__ __forceinline__ float siluf(float v) { return v / (1.f + __expf(-v)); }

// ---------------------------------------------------------------------------
// bf16 WMMA GEMM, fully padded (no bounds checks anywhere):
//   C[MPAD, NPAD] = A[MPAD, K] @ WT[K, NPAD]        (K % 32 == 0, NPAD % 32 == 0)
// WT is the pre-transposed weight (WT[k][n] = W[n][k]), zero-padded in n.
// Each wave computes a 32x32 C macro-tile: 2 A-frags x 2 B-frags -> 4 WMMAs
// per 32-wide k-step.  A-frag = 2x b128 loads/lane, B-frags = 32 scalar bf16 loads/lane.
// Grid is exact: (MPAD/32 * NPAD/32) / 8 blocks of 256 threads.
// ---------------------------------------------------------------------------
template <int K, int NPAD>
__global__ __launch_bounds__(256) void wmma_gemm_bf16(
    const __bf16* __restrict__ A, const __bf16* __restrict__ WT,
    float* __restrict__ C)
{
  constexpr int TILESN = NPAD / 32;
  const int lane = threadIdx.x & 31;
  const int wave = threadIdx.x >> 5;
  const int tileId = blockIdx.x * 8 + wave;
  const int tm = tileId / TILESN;
  const int tn = tileId % TILESN;
  const int m0 = tm * 32;
  const int n0 = tn * 32;
  const int half = lane >> 4;

  const __bf16* arow0 = A + (size_t)(m0 + (lane & 15)) * K;
  const __bf16* arow1 = arow0 + (size_t)16 * K;
  const __bf16* wbase = WT + n0;

  v8f acc00, acc01, acc10, acc11;
#pragma unroll
  for (int i = 0; i < 8; ++i) { acc00[i] = 0.f; acc01[i] = 0.f; acc10[i] = 0.f; acc11[i] = 0.f; }

#pragma unroll 2
  for (int k0 = 0; k0 < K; k0 += 32) {
    const int ka = k0 + 8 * half;
    v8bf a0lo = *(const v8bf*)(arow0 + ka);
    v8bf a0hi = *(const v8bf*)(arow0 + ka + 16);
    v8bf a1lo = *(const v8bf*)(arow1 + ka);
    v8bf a1hi = *(const v8bf*)(arow1 + ka + 16);
    const __bf16* wp = wbase + (lane & 15);
    v16bf b0, b1;
#pragma unroll
    for (int s = 0; s < 16; ++s) {
      const int kk = k0 + ((s < 8) ? (8 * half + s) : (16 + 8 * half + (s - 8)));
      b0[s] = wp[(size_t)kk * NPAD];
      b1[s] = wp[(size_t)kk * NPAD + 16];
    }
    if (k0 + 32 < K) __builtin_prefetch(arow0 + ka + 32, 0, 3);

    v16bf a0 = __builtin_shufflevector(a0lo, a0hi, 0,1,2,3,4,5,6,7,8,9,10,11,12,13,14,15);
    v16bf a1 = __builtin_shufflevector(a1lo, a1hi, 0,1,2,3,4,5,6,7,8,9,10,11,12,13,14,15);


    acc00 = __builtin_amdgcn_wmma_f32_16x16x32_bf16(false, a0, false, b0, (short)0, acc00, false, false);
    acc01 = __builtin_amdgcn_wmma_f32_16x16x32_bf16(false, a0, false, b1, (short)0, acc01, false, false);
    acc10 = __builtin_amdgcn_wmma_f32_16x16x32_bf16(false, a1, false, b0, (short)0, acc10, false, false);
    acc11 = __builtin_amdgcn_wmma_f32_16x16x32_bf16(false, a1, false, b1, (short)0, acc11, false, false);
  }

  // C layout: VGPR r -> M = r + 8*half, lane -> N = lane&15
  const int nc = n0 + (lane & 15);
  float* c0 = C + (size_t)(m0 + 8 * half) * NPAD + nc;
  float* c1 = c0 + (size_t)16 * NPAD;
#pragma unroll
  for (int r = 0; r < 8; ++r) {
    c0[(size_t)r * NPAD]      = acc00[r];
    c0[(size_t)r * NPAD + 16] = acc01[r];
    c1[(size_t)r * NPAD]      = acc10[r];
    c1[(size_t)r * NPAD + 16] = acc11[r];
  }
}

// ---------------------------------------------------------------------------
// fconv: per-(b,t) slice: LN over H, grouped conv over F (SAME, k=5, g=8),
// PReLU, residual.  xout = xin + fconv(xin).  In-place safe (block-local).
// ---------------------------------------------------------------------------
__global__ __launch_bounds__(256) void fconv_kernel(
    const float* __restrict__ xin, float* __restrict__ xout,
    const float* __restrict__ lw, const float* __restrict__ lb,
    const float* __restrict__ cw, const float* __restrict__ cb,
    const float* __restrict__ pa)
{
  __shared__ float ns[H_][F_ + 4];          // padded for conv halo
  const int bt = blockIdx.x;
  const int b = bt / T_, t = bt % T_;
  const int tid = threadIdx.x;

  for (int i = tid; i < F_ * H_; i += 256) {
    int f = i / H_, h = i % H_;
    ns[h][f + 2] = xin[(((size_t)b*F_ + f)*T_ + t)*H_ + h];
  }
  for (int i = tid; i < H_ * 4; i += 256) {
    int h = i >> 2, c = i & 3;
    ns[h][(c < 2) ? c : (F_ + c)] = 0.f;    // halo columns 0,1,131,132
  }
  __syncthreads();

  if (tid < F_) {                            // LN over H per f-column
    float s = 0.f, sq = 0.f;
    for (int h = 0; h < H_; ++h) { float v = ns[h][tid+2]; s += v; sq += v*v; }
    float m = s / H_;
    float rs = rsqrtf(sq / H_ - m*m + EPS_);
    for (int h = 0; h < H_; ++h) {
      float v = ns[h][tid+2];
      ns[h][tid+2] = (v - m) * rs * lw[h] + lb[h];
    }
  }
  __syncthreads();

  for (int i = tid; i < F_ * H_; i += 256) {
    int f = i / H_, o = i % H_;
    int g = o / CPG_;
    float a = cb[o];
#pragma unroll
    for (int k = 0; k < KW_; ++k)
      for (int ic = 0; ic < CPG_; ++ic)
        a += ns[g*CPG_ + ic][f + k] * cw[(o*CPG_ + ic)*KW_ + k];
    float v = (a >= 0.f) ? a : pa[o] * a;
    size_t idx = (((size_t)b*F_ + f)*T_ + t)*H_ + o;
    xout[idx] = xin[idx] + v;
  }
}

// ---------------------------------------------------------------------------
// full_block: per-(b,t) slice: LN(H) -> squeeze(S=8) -> silu -> F x F mix per
// group -> unsqueeze -> silu -> residual.  In-place.
// ---------------------------------------------------------------------------
__global__ __launch_bounds__(256) void fullblock_kernel(
    float* __restrict__ x,
    const float* __restrict__ nfw, const float* __restrict__ nfb,
    const float* __restrict__ sqw, const float* __restrict__ sqb,
    const float* __restrict__ fullw, const float* __restrict__ fullb,
    const float* __restrict__ usqw, const float* __restrict__ usqb)
{
  __shared__ float xs[F_][H_];
  __shared__ float s1[S_][F_];
  __shared__ float s2[S_][F_];
  const int bt = blockIdx.x;
  const int b = bt / T_, t = bt % T_;
  const int tid = threadIdx.x;

  for (int i = tid; i < F_ * H_; i += 256) {
    int f = i / H_, h = i % H_;
    xs[f][h] = x[(((size_t)b*F_ + f)*T_ + t)*H_ + h];
  }
  __syncthreads();

  if (tid < F_) {
    float s = 0.f, sq = 0.f;
    for (int h = 0; h < H_; ++h) { float v = xs[tid][h]; s += v; sq += v*v; }
    float m = s / H_;
    float rs = rsqrtf(sq / H_ - m*m + EPS_);
    float accs[S_];
#pragma unroll
    for (int si = 0; si < S_; ++si) accs[si] = 0.f;
    for (int h = 0; h < H_; ++h) {
      float yv = (xs[tid][h] - m) * rs * nfw[h] + nfb[h];
#pragma unroll
      for (int si = 0; si < S_; ++si) accs[si] += yv * sqw[si*H_ + h];
    }
#pragma unroll
    for (int si = 0; si < S_; ++si) s1[si][tid] = siluf(accs[si] + sqb[si]);
  }
  __syncthreads();

  for (int i = tid; i < S_ * F_; i += 256) {
    int g = i / F_, k = i % F_;
    float a = fullb[g*F_ + k];
    const float* wr = fullw + ((size_t)g*F_ + k)*F_;
    for (int h = 0; h < F_; ++h) a += s1[g][h] * wr[h];
    s2[g][k] = a;
  }
  __syncthreads();

  for (int i = tid; i < F_ * H_; i += 256) {
    int f = i / H_, ho = i % H_;
    float a = usqb[ho];
#pragma unroll
    for (int si = 0; si < S_; ++si) a += s2[si][f] * usqw[ho*S_ + si];
    size_t idx = (((size_t)b*F_ + f)*T_ + t)*H_ + ho;
    x[idx] += siluf(a);
  }
}

// ---------------------------------------------------------------------------
// LN over H, write bf16 row for WMMA GEMM A-operand.  rev flips T index.
// ---------------------------------------------------------------------------
__global__ __launch_bounds__(256) void ln_bf16_kernel(
    const float* __restrict__ x, const float* __restrict__ w,
    const float* __restrict__ bias, __bf16* __restrict__ out, int rev)
{
  int m = blockIdx.x * 256 + threadIdx.x;
  if (m >= MROWS) return;
  int n = m / T_, l = m % T_;
  int tsrc = rev ? (T_ - 1 - l) : l;
  const float* row = x + ((size_t)n*T_ + tsrc)*H_;
  float s = 0.f, sq = 0.f;
  for (int h = 0; h < H_; ++h) { float v = row[h]; s += v; sq += v*v; }
  float mean = s / H_;
  float rs = rsqrtf(sq / H_ - mean*mean + EPS_);
  __bf16* orow = out + (size_t)m * H_;
  for (int h = 0; h < H_; ++h)
    orow[h] = f2bf((row[h] - mean) * rs * w[h] + bias[h]);
}

// ---------------------------------------------------------------------------
// Depthwise causal conv (K=4, left pad 3) over xi = xz[:, :192], then silu.
// Writes f32 (for scan) and bf16 (for the x_w GEMM).
// ---------------------------------------------------------------------------
__global__ __launch_bounds__(256) void causal_conv_silu_kernel(
    const float* __restrict__ xz, const float* __restrict__ cw,
    const float* __restrict__ cb, float* __restrict__ xc,
    __bf16* __restrict__ xcbf)
{
  long tid = (long)blockIdx.x * 256 + threadIdx.x;
  if (tid >= (long)MROWS * DI_) return;
  int  c = (int)(tid % DI_);
  long m = tid / DI_;
  int  l = (int)(m % T_);
  long nbase = m - l;                         // n*T_
  float a = cb[c];
#pragma unroll
  for (int k = 0; k < KC_; ++k) {
    int ls = l - (KC_ - 1) + k;
    if (ls >= 0) a += xz[(size_t)(nbase + ls) * (2*DI_) + c] * cw[c*KC_ + k];
  }
  float v = siluf(a);
  xc[tid]   = v;
  xcbf[tid] = f2bf(v);
}

// ---------------------------------------------------------------------------
// Selective scan: one block per sequence n, one thread per channel d.
// dt = softplus(dbl[:, :6] @ dt_w^T + dt_b) computed inline.
// y = (scan + xc*D) * silu(z) -> bf16 for out-projection.
// dbl has padded row stride DBLS (=64): [0:6)=dt_r, [6:22)=B, [22:38)=C.
// ---------------------------------------------------------------------------
__global__ __launch_bounds__(192) void scan_kernel(
    const float* __restrict__ dbl, const float* __restrict__ xc,
    const float* __restrict__ xz,
    const float* __restrict__ dtw, const float* __restrict__ dtb,
    const float* __restrict__ Alog, const float* __restrict__ Dp,
    __bf16* __restrict__ ybf)
{
  __shared__ float sB[DS_], sC[DS_], sR[DR_];
  const int n = blockIdx.x;
  const int d = threadIdx.x;
  float Ad[DS_];
#pragma unroll
  for (int s = 0; s < DS_; ++s) Ad[s] = -__expf(Alog[d*DS_ + s]);
  float wr[DR_];
#pragma unroll
  for (int r = 0; r < DR_; ++r) wr[r] = dtw[d*DR_ + r];
  const float db = dtb[d], Dv = Dp[d];
  float h[DS_];
#pragma unroll
  for (int s = 0; s < DS_; ++s) h[s] = 0.f;

  for (int l = 0; l < T_; ++l) {
    size_t m = (size_t)n * T_ + l;
    if (d < DS_)              sB[d]         = dbl[m*DBLS + DR_ + d];
    else if (d < 2*DS_)       sC[d - DS_]   = dbl[m*DBLS + DR_ + DS_ + (d - DS_)];
    else if (d < 2*DS_ + DR_) sR[d - 2*DS_] = dbl[m*DBLS + (d - 2*DS_)];
    __syncthreads();
    float ta = db;
#pragma unroll
    for (int r = 0; r < DR_; ++r) ta += sR[r] * wr[r];
    float dtv = (ta > 20.f) ? ta : log1pf(__expf(ta));
    float xcv = xc[m*DI_ + d];
    float da = dtv * xcv;
    float y = 0.f;
#pragma unroll
    for (int s = 0; s < DS_; ++s) {
      h[s] = __expf(dtv * Ad[s]) * h[s] + da * sB[s];
      y += h[s] * sC[s];
    }
    y += xcv * Dv;
    float zv = xz[m*(2*DI_) + DI_ + d];
    y *= siluf(zv);
    ybf[m*DI_ + d] = f2bf(y);
    __syncthreads();
  }
}

__global__ __launch_bounds__(256) void residual_add_kernel(
    const float* __restrict__ outp, float* __restrict__ x, int rev)
{
  long tid = (long)blockIdx.x * 256 + threadIdx.x;
  if (tid >= (long)MROWS * H_) return;
  int  h = (int)(tid % H_);
  long m = tid / H_;
  int  l = (int)(m % T_);
  long n = m / T_;
  int tdst = rev ? (T_ - 1 - l) : l;
  x[((size_t)n*T_ + tdst)*H_ + h] += outp[(size_t)m * H_ + h];
}

// Convert W[N,K] f32 -> WT[K,NPAD] bf16 (transposed, zero-padded in n).
__global__ __launch_bounds__(256) void convert_transpose_w_kernel(
    const float* __restrict__ W, __bf16* __restrict__ WT, int N, int K, int Npad)
{
  int i = blockIdx.x * 256 + threadIdx.x;
  if (i >= K * Npad) return;
  int k = i / Npad, n = i % Npad;
  WT[i] = (n < N) ? f2bf(W[(size_t)n * K + k])
                  : __builtin_bit_cast(__bf16, (unsigned short)0);
}

__global__ __launch_bounds__(256) void zero_bf16_kernel(__bf16* __restrict__ p, int n)
{
  int i = blockIdx.x * 256 + threadIdx.x;
  if (i < n) p[i] = __builtin_bit_cast(__bf16, (unsigned short)0);
}

// ---------------------------------------------------------------------------
extern "C" void kernel_launch(void* const* d_in, const int* in_sizes, int n_in,
                              void* d_out, int out_size, void* d_ws, size_t ws_size,
                              hipStream_t stream)
{
  // Input order: x, fc1(lw,lb,cw,cb,prelu), fc2(...), nf_w,nf_b, sq_w,sq_b,
  // full_w,full_b, usq_w,usq_b, nm_w,nm_b, nt_w,nt_b, mamba_f{9}, mamba_b{9}
  const float* xin = (const float*)d_in[0];
  const float* fc1[5] = {(const float*)d_in[1], (const float*)d_in[2],
                         (const float*)d_in[3], (const float*)d_in[4],
                         (const float*)d_in[5]};
  const float* fc2[5] = {(const float*)d_in[6], (const float*)d_in[7],
                         (const float*)d_in[8], (const float*)d_in[9],
                         (const float*)d_in[10]};
  const float* nfw = (const float*)d_in[11];
  const float* nfb = (const float*)d_in[12];
  const float* sqw = (const float*)d_in[13];
  const float* sqb = (const float*)d_in[14];
  const float* fullw = (const float*)d_in[15];
  const float* fullb = (const float*)d_in[16];
  const float* usqw = (const float*)d_in[17];
  const float* usqb = (const float*)d_in[18];
  const float* nmw = (const float*)d_in[19];
  const float* nmb = (const float*)d_in[20];
  const float* ntw = (const float*)d_in[21];
  const float* ntb = (const float*)d_in[22];
  struct MP { const float *in_w,*conv_w,*conv_b,*x_w,*dt_w,*dt_b,*A_log,*D,*out_w; };
  MP mf = {(const float*)d_in[23],(const float*)d_in[24],(const float*)d_in[25],
           (const float*)d_in[26],(const float*)d_in[27],(const float*)d_in[28],
           (const float*)d_in[29],(const float*)d_in[30],(const float*)d_in[31]};
  MP mb = {(const float*)d_in[32],(const float*)d_in[33],(const float*)d_in[34],
           (const float*)d_in[35],(const float*)d_in[36],(const float*)d_in[37],
           (const float*)d_in[38],(const float*)d_in[39],(const float*)d_in[40]};

  char* ws = (char*)d_ws;
  size_t off = 0;
  auto alloc = [&](size_t bytes) -> char* {
    char* p = ws + off;
    off = (off + bytes + 255) & ~(size_t)255;
    return p;
  };
  // Transposed, padded bf16 weights: WT[K, NPAD]
  __bf16* wt_in_f  = (__bf16*)alloc((size_t)H_  * 384 * 2);   // 96 x 384
  __bf16* wt_x_f   = (__bf16*)alloc((size_t)DI_ * 64  * 2);   // 192 x 64
  __bf16* wt_out_f = (__bf16*)alloc((size_t)DI_ * 96  * 2);   // 192 x 96
  __bf16* wt_in_b  = (__bf16*)alloc((size_t)H_  * 384 * 2);
  __bf16* wt_x_b   = (__bf16*)alloc((size_t)DI_ * 64  * 2);
  __bf16* wt_out_b = (__bf16*)alloc((size_t)DI_ * 96  * 2);
  __bf16* Abf  = (__bf16*)alloc((size_t)MPAD * H_  * 2);   // LN output (A of gemm1)
  float*  xz   = (float*) alloc((size_t)MPAD * 384 * 4);   // in-projection output
  float*  xc   = (float*) alloc((size_t)MPAD * DI_ * 4);   // conv+silu f32 (reused as outp)
  __bf16* xcbf = (__bf16*)alloc((size_t)MPAD * DI_ * 2);   // A of gemm2
  float*  dblb = (float*) alloc((size_t)MPAD * DBLS * 4);  // x_w projection output
  __bf16* ybf  = (__bf16*)alloc((size_t)MPAD * DI_ * 2);   // A of gemm3
  float*  outp = xc;                                        // reuse after scan

  auto cvtT = [&](const float* src, __bf16* dst, int N, int K, int Npad) {
    int n = K * Npad;
    convert_transpose_w_kernel<<<(n + 255)/256, 256, 0, stream>>>(src, dst, N, K, Npad);
  };
  cvtT(mf.in_w,  wt_in_f,  2*DI_, H_,  384);
  cvtT(mf.x_w,   wt_x_f,   38,    DI_, 64);
  cvtT(mf.out_w, wt_out_f, H_,    DI_, 96);
  cvtT(mb.in_w,  wt_in_b,  2*DI_, H_,  384);
  cvtT(mb.x_w,   wt_x_b,   38,    DI_, 64);
  cvtT(mb.out_w, wt_out_b, H_,    DI_, 96);
  // Zero the pad rows of the GEMM A-operands (MROWS..MPAD).
  zero_bf16_kernel<<<((MPAD-MROWS)*H_  + 255)/256, 256, 0, stream>>>(Abf  + (size_t)MROWS*H_,  (MPAD-MROWS)*H_);
  zero_bf16_kernel<<<((MPAD-MROWS)*DI_ + 255)/256, 256, 0, stream>>>(xcbf + (size_t)MROWS*DI_, (MPAD-MROWS)*DI_);
  zero_bf16_kernel<<<((MPAD-MROWS)*DI_ + 255)/256, 256, 0, stream>>>(ybf  + (size_t)MROWS*DI_, (MPAD-MROWS)*DI_);

  float* xo = (float*)d_out;

  // Stage 1: x = x + fconv1(x)   (d_in -> d_out)
  fconv_kernel<<<B_*T_, 256, 0, stream>>>(xin, xo, fc1[0], fc1[1], fc1[2], fc1[3], fc1[4]);
  // Stage 2: x += full_block(x)  (in-place)
  fullblock_kernel<<<B_*T_, 256, 0, stream>>>(xo, nfw, nfb, sqw, sqb, fullw, fullb, usqw, usqb);
  // Stage 3: x += fconv2(x)      (in-place)
  fconv_kernel<<<B_*T_, 256, 0, stream>>>(xo, xo, fc2[0], fc2[1], fc2[2], fc2[3], fc2[4]);

  auto run_mamba = [&](const MP& mp, const float* nw, const float* nb,
                       __bf16* win, __bf16* wx, __bf16* wout, int rev) {
    ln_bf16_kernel<<<(MROWS + 255)/256, 256, 0, stream>>>(xo, nw, nb, Abf, rev);
    // gemm1: xz[MPAD,384] = Abf[MPAD,96] @ in_w^T     grid = 2016*12/8
    wmma_gemm_bf16<96, 384><<<(MPAD/32)*(384/32)/8, 256, 0, stream>>>(Abf, win, xz);
    {
      long tot = (long)MROWS * DI_;
      causal_conv_silu_kernel<<<(int)((tot + 255)/256), 256, 0, stream>>>(
          xz, mp.conv_w, mp.conv_b, xc, xcbf);
    }
    // gemm2: dbl[MPAD,64] = xcbf[MPAD,192] @ x_w^T    grid = 2016*2/8
    wmma_gemm_bf16<192, 64><<<(MPAD/32)*(64/32)/8, 256, 0, stream>>>(xcbf, wx, dblb);
    scan_kernel<<<NSEQ, 192, 0, stream>>>(dblb, xc, xz, mp.dt_w, mp.dt_b,
                                          mp.A_log, mp.D, ybf);
    // gemm3: outp[MPAD,96] = ybf[MPAD,192] @ out_w^T  grid = 2016*3/8
    wmma_gemm_bf16<192, 96><<<(MPAD/32)*(96/32)/8, 256, 0, stream>>>(ybf, wout, outp);
    {
      long tot = (long)MROWS * H_;
      residual_add_kernel<<<(int)((tot + 255)/256), 256, 0, stream>>>(outp, xo, rev);
    }
  };
  // Stage 4: forward mamba block
  run_mamba(mf, nmw, nmb, wt_in_f, wt_x_f, wt_out_f, 0);
  // Stage 5: backward (time-flipped) mamba block
  run_mamba(mb, ntw, ntb, wt_in_b, wt_x_b, wt_out_b, 1);
}
